// Seq2Seq_25288767439206
// MI455X (gfx1250) — matrix operand order repaired and checked by reading, not compiled, not run on hardware
//
#include <hip/hip_runtime.h>
#include <hip/hip_bf16.h>

// Problem constants (from reference)
#define B_  64
#define T_  80
#define IN_ 4096
#define H_  256
#define E_  256
#define V_  2880
#define L_  20
#define G4_ 1024   // 4*H

typedef __attribute__((ext_vector_type(16))) __bf16 v16bf;
typedef __attribute__((ext_vector_type(8)))  __bf16 v8bf;
typedef __attribute__((ext_vector_type(8)))  float  v8f;

#define GEMM_BIAS 1
#define GEMM_ACC  2

__device__ __forceinline__ float sigf(float x)     { return 1.f / (1.f + __expf(-x)); }
__device__ __forceinline__ float tanhfast(float x) { return 1.f - 2.f / (__expf(2.f * x) + 1.f); }

// ---- fp32 -> bf16 streaming conversion (8 elems/thread, b128 in / b128 out) ----
__global__ void cvt_f32_bf16(const float* __restrict__ in, __bf16* __restrict__ out,
                             long long n) {
  long long i = ((long long)blockIdx.x * blockDim.x + threadIdx.x) * 8;
  if (i >= n) return;
  float4 a = *(const float4*)(in + i);
  float4 b = *(const float4*)(in + i + 4);
  v8bf o;
  o[0] = (__bf16)a.x; o[1] = (__bf16)a.y; o[2] = (__bf16)a.z; o[3] = (__bf16)a.w;
  o[4] = (__bf16)b.x; o[5] = (__bf16)b.y; o[6] = (__bf16)b.z; o[7] = (__bf16)b.w;
  *(v8bf*)(out + i) = o;
}

// A fragment 16x32 bf16: lane l,h; elements 0..7 at K=k0+8h, 8..15 at K=k0+16+8h
__device__ __forceinline__ v16bf load_a_frag(const __bf16* row, int k0, int h) {
  v8bf lo = *(const v8bf*)(row + k0 + 8 * h);
  v8bf hi = *(const v8bf*)(row + k0 + 16 + 8 * h);
  return __builtin_shufflevector(lo, hi, 0, 1, 2, 3, 4, 5, 6, 7,
                                 8, 9, 10, 11, 12, 13, 14, 15);
}

// B fragment 32x16 bf16: lane l,h; elements 0..7 at K=k0+8h, 8..15 at K=k0+16+8h
__device__ __forceinline__ v16bf load_b_frag(const __bf16* row, int k0, int h) {
  v8bf lo = *(const v8bf*)(row + k0 + 8 * h);
  v8bf hi = *(const v8bf*)(row + k0 + 16 + 8 * h);
  return __builtin_shufflevector(lo, hi, 0, 1, 2, 3, 4, 5, 6, 7,
                                 8, 9, 10, 11, 12, 13, 14, 15);
}

// C[m,n] = sum_k A[m,k] * Bw[n,k]  (+bias[n]) (+C[m,n] if ACC)
// A: M x K row-major (bf16), Bw: N x K row-major (bf16 weight (out,in)).
// Requires M%16==0, N%(16*NB)==0, K%32==0.
// Each wave: 1 M-tile x NB N-tiles (A fragment loaded once per K-step, NB WMMAs).
template <int NB>
__global__ __launch_bounds__(256) void gemm_wmma_bf16(
    const __bf16* __restrict__ A, int lda,
    const __bf16* __restrict__ Bw, int ldb,
    const float* __restrict__ bias,
    float* __restrict__ C, long long ldc,
    int Nwt, int K, int flags)
{
  const int wave = threadIdx.x >> 5;
  const int lane = threadIdx.x & 31;
  const int tg = blockIdx.x * 8 + wave;   // group of NB consecutive N-tiles
  if (tg >= Nwt) return;                  // wave-uniform: EXEC all-ones for WMMA
  const int tm = blockIdx.y;
  const int l = lane & 15;
  const int h = lane >> 4;

  const __bf16* arow = A + (long long)(tm * 16 + l) * lda;
  const __bf16* brow[NB];
#pragma unroll
  for (int nb = 0; nb < NB; ++nb)
    brow[nb] = Bw + (long long)((tg * NB + nb) * 16 + l) * ldb;

  v8f acc[NB];
#pragma unroll
  for (int nb = 0; nb < NB; ++nb)
    acc[nb] = (v8f){0.f, 0.f, 0.f, 0.f, 0.f, 0.f, 0.f, 0.f};

  for (int k0 = 0; k0 < K; k0 += 32) {
    if (k0 + 32 < K) {
      __builtin_prefetch(arow + k0 + 32 + 8 * h, 0, 1);   // global_prefetch_b8
#pragma unroll
      for (int nb = 0; nb < NB; ++nb)
        __builtin_prefetch(brow[nb] + k0 + 32 + 16 * h, 0, 1);
    }
    v16bf av = load_a_frag(arow, k0, h);
    v16bf bv[NB];
#pragma unroll
    for (int nb = 0; nb < NB; ++nb) bv[nb] = load_b_frag(brow[nb], k0, h);
#pragma unroll
    for (int nb = 0; nb < NB; ++nb)
      acc[nb] = __builtin_amdgcn_wmma_f32_16x16x32_bf16(false, av, false, bv[nb],
                                                        (short)0, acc[nb], false, false);
  }

#pragma unroll
  for (int nb = 0; nb < NB; ++nb) {
    const int col = (tg * NB + nb) * 16 + l;
    const float bb = (flags & GEMM_BIAS) ? bias[col] : 0.f;
#pragma unroll
    for (int r = 0; r < 8; ++r) {
      const long long row = (long long)(tm * 16 + 8 * h + r);
      float* cp = C + row * ldc + col;
      float v = acc[nb][r] + bb;
      if (flags & GEMM_ACC) v += *cp;
      *cp = v;
    }
  }
}

// g = Gx(+Gh); i,f,g,o gates; in-place h,c update (fp32); bf16 h copy for next GEMM;
// optional fp32 ys and/or bf16 ys sequence outputs.
__global__ void lstm_gate(const float* __restrict__ Gx, long long gxStride,
                          const float* __restrict__ Gh,
                          float* __restrict__ c, float* __restrict__ h,
                          __bf16* __restrict__ h_bf,
                          float* __restrict__ ys, long long ysStride,
                          __bf16* __restrict__ ys_bf, long long ysbfStride)
{
  int i = blockIdx.x * blockDim.x + threadIdx.x;
  if (i >= B_ * H_) return;
  int b = i / H_, n = i % H_;
  const float* gx = Gx + (long long)b * gxStride;
  float gi = gx[n], gf = gx[n + H_], gg = gx[n + 2 * H_], go = gx[n + 3 * H_];
  if (Gh) {
    const float* gh = Gh + (long long)b * G4_;
    gi += gh[n]; gf += gh[n + H_]; gg += gh[n + 2 * H_]; go += gh[n + 3 * H_];
  }
  float cc = sigf(gf) * c[i] + sigf(gi) * tanhfast(gg);
  float hh = sigf(go) * tanhfast(cc);
  c[i] = cc; h[i] = hh; h_bf[i] = (__bf16)hh;
  if (ys)    ys[(long long)b * ysStride + n] = hh;
  if (ys_bf) ys_bf[(long long)b * ysbfStride + n] = (__bf16)hh;
}

__global__ void init_state(float* h1, float* c1, float* h2, float* c2,
                           __bf16* h1_bf, __bf16* h2_bf,
                           float* emb, const int* __restrict__ bos,
                           const float* __restrict__ embed)
{
  int i = blockIdx.x * blockDim.x + threadIdx.x;
  if (i >= B_ * H_) return;
  h1[i] = 0.f; c1[i] = 0.f; h2[i] = 0.f; c2[i] = 0.f;
  h1_bf[i] = (__bf16)0.f; h2_bf[i] = (__bf16)0.f;
  int b = i / E_, e = i % E_;
  emb[i] = embed[(long long)bos[b] * E_ + e];
}

__global__ void attn_scores(const float* __restrict__ h1,
                            const float* __restrict__ enc,
                            float* __restrict__ sc)
{
  int i = blockIdx.x * blockDim.x + threadIdx.x;   // i = b*T + t
  if (i >= B_ * T_) return;
  int b = i / T_;
  const float4* hp = (const float4*)(h1 + (long long)b * H_);
  const float4* ep = (const float4*)(enc + (long long)i * H_);
  float s = 0.f;
#pragma unroll 4
  for (int k = 0; k < H_ / 4; ++k) {
    float4 a = hp[k], e = ep[k];
    s += a.x * e.x + a.y * e.y + a.z * e.z + a.w * e.w;
  }
  sc[i] = s;
}

__global__ void softmax_T(float* __restrict__ sc) {
  int b = blockIdx.x * blockDim.x + threadIdx.x;
  if (b >= B_) return;
  float* p = sc + (long long)b * T_;
  float m = -3.4e38f;
  for (int t = 0; t < T_; ++t) m = fmaxf(m, p[t]);
  float s = 0.f;
  for (int t = 0; t < T_; ++t) { float e = __expf(p[t] - m); p[t] = e; s += e; }
  float inv = 1.f / s;
  for (int t = 0; t < T_; ++t) p[t] *= inv;
}

// ctx in fp32 accumulate, emitted directly as bf16 (only consumer is a GEMM)
__global__ void attn_ctx(const float* __restrict__ sc,
                         const float* __restrict__ enc,
                         __bf16* __restrict__ ctx_bf)
{
  int i = blockIdx.x * blockDim.x + threadIdx.x;   // i = b*H + h
  if (i >= B_ * H_) return;
  int b = i / H_, hh = i % H_;
  float s = 0.f;
  for (int t = 0; t < T_; ++t)
    s += sc[b * T_ + t] * enc[((long long)b * T_ + t) * H_ + hh];
  ctx_bf[i] = (__bf16)s;
}

__global__ void concat_emb_h(const float* __restrict__ emb,
                             const float* __restrict__ h1,
                             __bf16* __restrict__ x2_bf)
{
  int i = blockIdx.x * blockDim.x + threadIdx.x;
  if (i >= B_ * (E_ + H_)) return;
  int b = i / (E_ + H_), j = i % (E_ + H_);
  float v = (j < E_) ? emb[b * E_ + j] : h1[b * H_ + j - E_];
  x2_bf[i] = (__bf16)v;
}

__global__ __launch_bounds__(256) void argmax_embed(
    const float* __restrict__ logits, long long rowStride,
    const float* __restrict__ embed, float* __restrict__ emb)
{
  __shared__ float sv[256];
  __shared__ int   si[256];
  int b = blockIdx.x, t = threadIdx.x;
  const float* p = logits + (long long)b * rowStride;
  float best = -3.4e38f; int bi = 0;
  for (int v = t; v < V_; v += 256) {
    float x = p[v];
    if (x > best) { best = x; bi = v; }
  }
  sv[t] = best; si[t] = bi;
  __syncthreads();
  for (int off = 128; off > 0; off >>= 1) {
    if (t < off) {
      if (sv[t + off] > sv[t] || (sv[t + off] == sv[t] && si[t + off] < si[t])) {
        sv[t] = sv[t + off]; si[t] = si[t + off];
      }
    }
    __syncthreads();
  }
  int idx = si[0];
  emb[(long long)b * E_ + t] = embed[(long long)idx * E_ + t];  // E_==256==blockDim
}

extern "C" void kernel_launch(void* const* d_in, const int* in_sizes, int n_in,
                              void* d_out, int out_size, void* d_ws, size_t ws_size,
                              hipStream_t stream) {
  (void)in_sizes; (void)n_in; (void)out_size; (void)ws_size;
  const float* src   = (const float*)d_in[0];
  const int*   bos   = (const int*)  d_in[1];
  const float* embed = (const float*)d_in[2];
  const float* e1Wih = (const float*)d_in[3];
  const float* e1Whh = (const float*)d_in[4];
  const float* e1b   = (const float*)d_in[5];
  const float* e2Wih = (const float*)d_in[6];
  const float* e2Whh = (const float*)d_in[7];
  const float* e2b   = (const float*)d_in[8];
  const float* d1Wih = (const float*)d_in[9];
  const float* d1Whh = (const float*)d_in[10];
  const float* d1b   = (const float*)d_in[11];
  const float* d2Wih = (const float*)d_in[12];
  const float* d2Whh = (const float*)d_in[13];
  const float* d2b   = (const float*)d_in[14];
  const float* outW  = (const float*)d_in[15];
  const float* outb  = (const float*)d_in[16];
  float* out = (float*)d_out;

  // ---------------- Workspace layout ----------------
  // fp32 region
  float* G   = (float*)d_ws;                     // 5120*1024 (shared by both enc layers)
  float* enc = G   + (long long)B_ * T_ * G4_;   // 5120*256 fp32 (attention operand)
  float* Gs  = enc + (long long)B_ * T_ * H_;    // 64*1024 step gates
  float* h1  = Gs  + (long long)B_ * G4_;
  float* c1  = h1 + B_ * H_;
  float* h2  = c1 + B_ * H_;
  float* c2  = h2 + B_ * H_;
  float* emb = c2 + B_ * H_;                     // 64*256
  float* sc  = emb + B_ * E_;                    // 64*80
  float* f32_end = sc + B_ * T_;
  // bf16 region
  __bf16* bfp     = (__bf16*)f32_end;
  __bf16* src_bf  = bfp;                    bfp += (long long)B_ * T_ * IN_;  // 20.97M
  __bf16* e1Wih_b = bfp;                    bfp += (long long)G4_ * IN_;
  __bf16* e1Whh_b = bfp;                    bfp += (long long)G4_ * H_;
  __bf16* e2Wih_b = bfp;                    bfp += (long long)G4_ * H_;
  __bf16* e2Whh_b = bfp;                    bfp += (long long)G4_ * H_;
  __bf16* d1Wih_b = bfp;                    bfp += (long long)G4_ * H_;
  __bf16* d1Whh_b = bfp;                    bfp += (long long)G4_ * H_;
  __bf16* d2Wih_b = bfp;                    bfp += (long long)G4_ * (E_ + H_);
  __bf16* d2Whh_b = bfp;                    bfp += (long long)G4_ * H_;
  __bf16* outW_b  = bfp;                    bfp += (long long)V_ * H_;
  __bf16* mid_bf  = bfp;                    bfp += (long long)B_ * T_ * H_;   // enc1 outputs
  __bf16* h1_bf   = bfp;                    bfp += B_ * H_;
  __bf16* h2_bf   = bfp;                    bfp += B_ * H_;
  __bf16* ctx_bf  = bfp;                    bfp += B_ * H_;
  __bf16* x2_bf   = bfp;                    bfp += B_ * (E_ + H_);

  auto cvt = [&](const float* in, __bf16* o, long long n) {
    cvt_f32_bf16<<<(unsigned)((n / 8 + 255) / 256), 256, 0, stream>>>(in, o, n);
  };
  // Big GEMMs (M large): NB=4 (A fragment reused for 4 WMMAs)
  auto gemm4 = [&](const __bf16* A, int lda, const __bf16* Bw, int ldb,
                   const float* bias, float* C, long long ldc,
                   int M, int N, int K, int flags) {
    int Nwt = N / (16 * 4);
    dim3 grid((Nwt + 7) / 8, M / 16, 1);
    gemm_wmma_bf16<4><<<grid, 256, 0, stream>>>(A, lda, Bw, ldb, bias, C, ldc,
                                                Nwt, K, flags);
  };
  // Small GEMMs (M=64): NB=2 keeps more waves in flight
  auto gemm2 = [&](const __bf16* A, int lda, const __bf16* Bw, int ldb,
                   const float* bias, float* C, long long ldc,
                   int M, int N, int K, int flags) {
    int Nwt = N / (16 * 2);
    dim3 grid((Nwt + 7) / 8, M / 16, 1);
    gemm_wmma_bf16<2><<<grid, 256, 0, stream>>>(A, lda, Bw, ldb, bias, C, ldc,
                                                Nwt, K, flags);
  };

  // ---- One-time bf16 conversions (streaming, HBM-bound: ~60 MB total) ----
  cvt(src,   src_bf,  (long long)B_ * T_ * IN_);
  cvt(e1Wih, e1Wih_b, (long long)G4_ * IN_);
  cvt(e1Whh, e1Whh_b, (long long)G4_ * H_);
  cvt(e2Wih, e2Wih_b, (long long)G4_ * H_);
  cvt(e2Whh, e2Whh_b, (long long)G4_ * H_);
  cvt(d1Wih, d1Wih_b, (long long)G4_ * H_);
  cvt(d1Whh, d1Whh_b, (long long)G4_ * H_);
  cvt(d2Wih, d2Wih_b, (long long)G4_ * (E_ + H_));
  cvt(d2Whh, d2Whh_b, (long long)G4_ * H_);
  cvt(outW,  outW_b,  (long long)V_ * H_);

  // init h/c/emb
  init_state<<<(B_ * H_ + 255) / 256, 256, 0, stream>>>(h1, c1, h2, c2,
                                                        h1_bf, h2_bf, emb, bos, embed);

  // ---- Encoder layer 1 ----
  // Hoisted input transform: G[b*T+t, :] = src @ Wih.T + b  (43 GFLOP, all WMMA)
  gemm4(src_bf, IN_, e1Wih_b, IN_, e1b, G, G4_, B_ * T_, G4_, IN_, GEMM_BIAS);
  for (int t = 0; t < T_; ++t) {
    gemm2(h1_bf, H_, e1Whh_b, H_, nullptr, Gs, G4_, B_, G4_, H_, 0);
    lstm_gate<<<(B_ * H_ + 255) / 256, 256, 0, stream>>>(
        G + (long long)t * G4_, (long long)T_ * G4_, Gs, c1, h1, h1_bf,
        nullptr, 0LL, mid_bf + (long long)t * H_, (long long)T_ * H_);
  }

  // ---- Encoder layer 2 ----
  gemm4(mid_bf, H_, e2Wih_b, H_, e2b, G, G4_, B_ * T_, G4_, H_, GEMM_BIAS);
  for (int t = 0; t < T_; ++t) {
    gemm2(h2_bf, H_, e2Whh_b, H_, nullptr, Gs, G4_, B_, G4_, H_, 0);
    lstm_gate<<<(B_ * H_ + 255) / 256, 256, 0, stream>>>(
        G + (long long)t * G4_, (long long)T_ * G4_, Gs, c2, h2, h2_bf,
        enc + (long long)t * H_, (long long)T_ * H_, nullptr, 0LL);
  }

  // ---- Decoder (h1/c1, h2/c2 hold encoder final states) ----
  for (int s = 0; s < L_; ++s) {
    attn_scores<<<(B_ * T_ + 255) / 256, 256, 0, stream>>>(h1, enc, sc);
    softmax_T<<<1, 64, 0, stream>>>(sc);
    attn_ctx<<<(B_ * H_ + 255) / 256, 256, 0, stream>>>(sc, enc, ctx_bf);

    // dec1 cell
    gemm2(ctx_bf, H_, d1Wih_b, H_, d1b, Gs, G4_, B_, G4_, H_, GEMM_BIAS);
    gemm2(h1_bf, H_, d1Whh_b, H_, nullptr, Gs, G4_, B_, G4_, H_, GEMM_ACC);
    lstm_gate<<<(B_ * H_ + 255) / 256, 256, 0, stream>>>(
        Gs, (long long)G4_, nullptr, c1, h1, h1_bf, nullptr, 0LL, nullptr, 0LL);

    // dec2 cell on [emb, h1]
    concat_emb_h<<<(B_ * (E_ + H_) + 255) / 256, 256, 0, stream>>>(emb, h1, x2_bf);
    gemm2(x2_bf, E_ + H_, d2Wih_b, E_ + H_, d2b, Gs, G4_, B_, G4_, E_ + H_, GEMM_BIAS);
    gemm2(h2_bf, H_, d2Whh_b, H_, nullptr, Gs, G4_, B_, G4_, H_, GEMM_ACC);
    lstm_gate<<<(B_ * H_ + 255) / 256, 256, 0, stream>>>(
        Gs, (long long)G4_, nullptr, c2, h2, h2_bf, nullptr, 0LL, nullptr, 0LL);

    // logits straight into d_out[b, s, :] via strided C
    gemm2(h2_bf, H_, outW_b, H_, outb, out + (long long)s * V_, (long long)L_ * V_,
          B_, V_, H_, GEMM_BIAS);

    // greedy feedback
    argmax_embed<<<B_, 256, 0, stream>>>(out + (long long)s * V_,
                                         (long long)L_ * V_, embed, emb);
  }
}
